// DoubleStreamBlock_68710886801454
// MI455X (gfx1250) — compile-verified, matrix operand order checked by reading, not run on hardware
//
#include <hip/hip_runtime.h>
#include <math.h>

// ---------------------------------------------------------------------------
// Types for CDNA5 WMMA (gfx1250, wave32)
// ---------------------------------------------------------------------------
typedef __bf16 v16bf __attribute__((ext_vector_type(16)));
typedef float  v8f   __attribute__((ext_vector_type(8)));

#define HID     2048
#define HEADS   16
#define HD      128
#define L_TXT   256
#define L_IMG   2048
#define L_ALL   2304
#define NMOD    12288
#define QKV_N   6144
#define MLP_HID 8192

// ---------------------------------------------------------------------------
// CDNA5 async global->LDS staging (ASYNCcnt-tracked DMA path).
// clang signature (confirmed by diagnostics): 
//   __builtin_amdgcn_global_load_async_to_lds_b128(v4i AS1*, v4i AS3*, Ii, Ii)
// where v4i is a GCC vector of 4 ints (16 bytes).
// Falls back to a synchronous VGPR bounce if the builtin is unavailable.
// ---------------------------------------------------------------------------
#if defined(__has_builtin)
#  if __has_builtin(__builtin_amdgcn_global_load_async_to_lds_b128)
#    define ASYNC_LDS 1
#  endif
#endif
#ifndef ASYNC_LDS
#  define ASYNC_LDS 0
#endif

typedef int v4i __attribute__((vector_size(16)));
typedef __attribute__((address_space(1))) v4i as1_v4i;
typedef __attribute__((address_space(3))) v4i as3_v4i;

__device__ __forceinline__ void cp128_g2l(const void* g, void* l) {
#if ASYNC_LDS
  __builtin_amdgcn_global_load_async_to_lds_b128(
      (as1_v4i*)(void*)g, (as3_v4i*)l, 0, 0);
#else
  *(uint4*)l = *(const uint4*)g;
#endif
}

__device__ __forceinline__ void async_wait0() {
#if ASYNC_LDS
#  if __has_builtin(__builtin_amdgcn_s_wait_asynccnt)
  __builtin_amdgcn_s_wait_asynccnt(0);
#  else
  asm volatile("s_wait_asynccnt 0" ::: "memory");
#  endif
#endif
}

__device__ __forceinline__ unsigned short f2bf(float x) {
  unsigned int u = __float_as_uint(x);
  u += 0x7fffu + ((u >> 16) & 1u);           // round-to-nearest-even
  return (unsigned short)(u >> 16);
}

__device__ __forceinline__ float gelu_tanh(float x) {
  float x3 = x * x * x;
  return 0.5f * x * (1.0f + tanhf(0.79788456080286535588f * (x + 0.044715f * x3)));
}

// Build a 16x32 bf16 WMMA fragment (A layout; B uses the mirrored layout with
// n <-> m) from LDS where the 16-wide index selects a row of `stride` bf16
// elements and k is contiguous.  Lane l: idx = l&15, half = l>>4.
// VGPR0..3 hold k = half*8 + {0..7}; VGPR4..7 hold k = 16 + half*8 + {0..7}.
__device__ __forceinline__ v16bf load_frag(const unsigned short* p0, int stride, int lane) {
  union { v16bf v; unsigned int u[8]; } f;
  const unsigned int* p =
      (const unsigned int*)(p0 + (lane & 15) * stride + ((lane >> 4) << 3));
  f.u[0] = p[0];  f.u[1] = p[1];  f.u[2] = p[2];  f.u[3] = p[3];
  f.u[4] = p[8];  f.u[5] = p[9];  f.u[6] = p[10]; f.u[7] = p[11];
  return f.v;
}

__device__ __forceinline__ v8f wmma_bf16(v16bf a, v16bf b, v8f c) {
  return __builtin_amdgcn_wmma_f32_16x16x32_bf16(
      /*neg_a=*/false, a, /*neg_b=*/false, b,
      /*c_mod=*/(short)0, c, /*reuse_a=*/false, /*reuse_b=*/false);
}

__device__ __forceinline__ v8f zero8() {
  v8f z;
#pragma unroll
  for (int r = 0; r < 8; r++) z[r] = 0.0f;
  return z;
}

// ---------------------------------------------------------------------------
// Modulation: out[n] = dot(silu(vec), W[:,n]) + b[n]   (N = 12288)
// ---------------------------------------------------------------------------
__launch_bounds__(256)
__global__ void modulation_kernel(const float* __restrict__ vec,
                                  const float* __restrict__ w,
                                  const float* __restrict__ b,
                                  float* __restrict__ out) {
  __shared__ float sv[HID];
  const int tid = threadIdx.x;
  for (int i = tid; i < HID; i += 256) {
    float v = vec[i];
    sv[i] = v / (1.0f + __expf(-v));
  }
  __syncthreads();
  const int n = blockIdx.x * 256 + tid;
  float acc = 0.0f;
  for (int k = 0; k < HID; k++) acc = fmaf(sv[k], w[(size_t)k * NMOD + n], acc);
  out[n] = acc + b[n];
}

// ---------------------------------------------------------------------------
// LayerNorm + modulate + cast bf16: out = bf16((1+sc)*LN(x) + sh)
// one block per token row (2048 cols)
// ---------------------------------------------------------------------------
__launch_bounds__(256)
__global__ void ln_mod_kernel(const float* __restrict__ x,
                              const float* __restrict__ sh,
                              const float* __restrict__ sc,
                              unsigned short* __restrict__ out) {
  __shared__ float red[256];
  const int row = blockIdx.x, tid = threadIdx.x;
  const float* xr = x + (size_t)row * HID;
  float s = 0.0f, s2 = 0.0f;
  for (int i = tid; i < HID; i += 256) {
    float v = xr[i];
    s += v; s2 += v * v;
  }
  red[tid] = s; __syncthreads();
  for (int st = 128; st > 0; st >>= 1) {
    if (tid < st) red[tid] += red[tid + st];
    __syncthreads();
  }
  float mu = red[0] * (1.0f / HID);
  __syncthreads();
  red[tid] = s2; __syncthreads();
  for (int st = 128; st > 0; st >>= 1) {
    if (tid < st) red[tid] += red[tid + st];
    __syncthreads();
  }
  float var = red[0] * (1.0f / HID) - mu * mu;
  float rstd = rsqrtf(var + 1e-6f);
  unsigned short* orow = out + (size_t)row * HID;
  for (int i = tid; i < HID; i += 256) {
    float v = (xr[i] - mu) * rstd;
    orow[i] = f2bf((1.0f + sc[i]) * v + sh[i]);
  }
}

// ---------------------------------------------------------------------------
// Generic bf16-WMMA GEMM: C[M,N] = A[M,K](bf16) * W[K,N](f32->bf16) + bias
// block tile 128x128, K-step 32, 8 waves (wave32), wave tile 64x32.
// A tiles staged with global_load_async_to_lds; W converted f32->bf16 in regs.
// Epilogues: 0 = f32 out, 1 = bf16(gelu(out)), 2 = f32 res + gate*out
// ---------------------------------------------------------------------------
enum { EPI_F32 = 0, EPI_GELU_BF16 = 1, EPI_RESGATE_F32 = 2 };

__launch_bounds__(256)
__global__ void gemm_bf16w(const unsigned short* __restrict__ A,
                           const float* __restrict__ W,
                           const float* __restrict__ bias,
                           int M, int N, int K, int epi,
                           const float* __restrict__ res,
                           const float* __restrict__ gate,
                           float* __restrict__ outF,
                           unsigned short* __restrict__ outH) {
  __shared__ unsigned short As[128 * 32];   // [m][k]
  __shared__ unsigned short Bs[128 * 32];   // [n][k] (transposed on load)
  const int tid  = threadIdx.x;
  const int lane = tid & 31;
  const int wave = tid >> 5;
  const int wm   = (wave & 1) * 64;
  const int wn   = (wave >> 1) * 32;
  const int bm   = blockIdx.y * 128;
  const int bn   = blockIdx.x * 128;

  // thread-constant staging coordinates, hoisted out of the K loop
  const int arow = tid >> 2;                 // rows arow and arow+64
  const int ack  = (tid & 3) << 3;           // bf16 element offset in k
  const unsigned short* aSrc = A + (size_t)(bm + arow) * K + ack;
  unsigned short*       aDst = As + arow * 32 + ack;

  const int wn4 = (tid & 31) << 2;           // 4 consecutive n columns
  const float* wSrc = W + (size_t)wave * N + bn + wn4;  // rows wave + 8*i
  unsigned short* wDst = Bs + wn4 * 32 + wave;          // col k = wave + 8*i

  v8f acc[4][2];
#pragma unroll
  for (int i = 0; i < 4; i++)
#pragma unroll
    for (int j = 0; j < 2; j++) acc[i][j] = zero8();

  for (int k0 = 0; k0 < K; k0 += 32) {
    // stage A tile (8KB) via async DMA: 2 x b128 per thread
    cp128_g2l(aSrc + k0, aDst);
    cp128_g2l(aSrc + (size_t)64 * K + k0, aDst + 64 * 32);

    // stage W tile with f32->bf16 conversion and [k][n] -> [n][k] transpose
    const float* wp = wSrc + (size_t)k0 * N;
#pragma unroll
    for (int i = 0; i < 4; i++) {
      float4 w4 = *(const float4*)(wp + (size_t)(8 * i) * N);
      unsigned short* d = wDst + 8 * i;
      d[0 * 32] = f2bf(w4.x);
      d[1 * 32] = f2bf(w4.y);
      d[2 * 32] = f2bf(w4.z);
      d[3 * 32] = f2bf(w4.w);
    }
    if (k0 + 32 < K)
      __builtin_prefetch(wp + (size_t)32 * N, 0, 0);
    async_wait0();
    __syncthreads();

    v16bf b0 = load_frag(Bs + (wn +  0) * 32, 32, lane);
    v16bf b1 = load_frag(Bs + (wn + 16) * 32, 32, lane);
#pragma unroll
    for (int i = 0; i < 4; i++) {
      v16bf a = load_frag(As + (wm + i * 16) * 32, 32, lane);
      acc[i][0] = wmma_bf16(a, b0, acc[i][0]);
      acc[i][1] = wmma_bf16(a, b1, acc[i][1]);
    }
    __syncthreads();
  }

  // epilogue: C layout — VGPR r holds row (r + 8*(lane>>4)), col = lane&15
  const int half = lane >> 4;
  const int nl   = lane & 15;
#pragma unroll
  for (int i = 0; i < 4; i++) {
    int rowBase = bm + wm + i * 16 + half * 8;
#pragma unroll
    for (int j = 0; j < 2; j++) {
      int col = bn + wn + j * 16 + nl;
      float b = bias[col];
      float g = (epi == EPI_RESGATE_F32) ? gate[col] : 0.0f;
#pragma unroll
      for (int r = 0; r < 8; r++) {
        size_t idx = (size_t)(rowBase + r) * N + col;
        float v = acc[i][j][r] + b;
        if (epi == EPI_F32)            outF[idx] = v;
        else if (epi == EPI_GELU_BF16) outH[idx] = f2bf(gelu_tanh(v));
        else                           outF[idx] = res[idx] + g * v;
      }
    }
  }
}

// ---------------------------------------------------------------------------
// QKV prep: per (token l, head h): RMSNorm(q,k) * scale, RoPE(q,k),
// q *= 1/sqrt(128), cast bf16, scatter to [H][L][D] Q/K/V buffers.
// block = 128 threads (one per d), grid = (L_ALL, HEADS)
// ---------------------------------------------------------------------------
__launch_bounds__(128)
__global__ void qkv_prep_kernel(const float* __restrict__ qkvTxt,
                                const float* __restrict__ qkvImg,
                                const float* __restrict__ qsT,
                                const float* __restrict__ ksT,
                                const float* __restrict__ qsI,
                                const float* __restrict__ ksI,
                                const float* __restrict__ pe,
                                unsigned short* __restrict__ Qb,
                                unsigned short* __restrict__ Kb,
                                unsigned short* __restrict__ Vb) {
  __shared__ float red[128];
  __shared__ float shq[128];
  __shared__ float shk[128];
  const int l = blockIdx.x, h = blockIdx.y, d = threadIdx.x;
  const float *src, *qs, *ks;
  if (l < L_TXT) { src = qkvTxt + (size_t)l * QKV_N;            qs = qsT; ks = ksT; }
  else           { src = qkvImg + (size_t)(l - L_TXT) * QKV_N;  qs = qsI; ks = ksI; }
  float q = src[           h * HD + d];
  float k = src[HID      + h * HD + d];
  float v = src[2 * HID  + h * HD + d];

  red[d] = q * q; __syncthreads();
  for (int st = 64; st > 0; st >>= 1) { if (d < st) red[d] += red[d + st]; __syncthreads(); }
  float rq = rsqrtf(red[0] * (1.0f / HD) + 1e-6f);
  __syncthreads();
  red[d] = k * k; __syncthreads();
  for (int st = 64; st > 0; st >>= 1) { if (d < st) red[d] += red[d + st]; __syncthreads(); }
  float rk = rsqrtf(red[0] * (1.0f / HD) + 1e-6f);
  __syncthreads();

  shq[d] = q * rq * qs[d];
  shk[d] = k * rk * ks[d];
  __syncthreads();

  // RoPE: out[2*d0+j] = pe[l,d0,j,0]*x[2*d0] + pe[l,d0,j,1]*x[2*d0+1]
  const int d0 = d >> 1, j = d & 1;
  const float* pp = pe + (((size_t)l * (HD / 2) + d0) * 2 + j) * 2;
  float qo = pp[0] * shq[2 * d0] + pp[1] * shq[2 * d0 + 1];
  float ko = pp[0] * shk[2 * d0] + pp[1] * shk[2 * d0 + 1];
  qo *= 0.08838834764831845f;   // 1/sqrt(128) folded into q

  size_t base = ((size_t)h * L_ALL + l) * HD + d;
  Qb[base] = f2bf(qo);
  Kb[base] = f2bf(ko);
  Vb[base] = f2bf(v);
}

// ---------------------------------------------------------------------------
// Flash attention: grid (L_ALL/128, HEADS), 256 threads (8 waves).
// Each wave owns 16 query rows; K/V staged 32 keys at a time.
// Q/K tiles staged with global_load_async_to_lds; V transposed in regs.
// S = Q K^T (WMMA), online softmax (shfl reductions in 16-lane halves),
// P re-laid out via per-wave LDS scratch, O += P V (WMMA).
// ---------------------------------------------------------------------------
__launch_bounds__(256)
__global__ void attn_kernel(const unsigned short* __restrict__ Qg,
                            const unsigned short* __restrict__ Kg,
                            const unsigned short* __restrict__ Vg,
                            unsigned short* __restrict__ Out) {
  __shared__ unsigned short Qs[128 * 128];   // [m][d]      32 KB
  __shared__ unsigned short Ks[32 * 128];    // [key][d]     8 KB
  __shared__ unsigned short Vs[128 * 32];    // [d][key]     8 KB
  __shared__ unsigned short Ps[8][16 * 32];  // per-wave P   8 KB
  const int tid = threadIdx.x, lane = tid & 31, wave = tid >> 5;
  const int half = lane >> 4, nl = lane & 15;
  const int h  = blockIdx.y;
  const int q0 = blockIdx.x * 128;
  const size_t hb = (size_t)h * L_ALL * HD;

  // thread-constant staging coordinates
  const int srow = tid >> 4;                 // + 16*i
  const int sck  = (tid & 15) << 3;          // element offset

  // Q tile: 8 x b128 per thread, async
  {
    const unsigned short* qsrc = Qg + hb + (size_t)(q0 + srow) * HD + sck;
    unsigned short* qdst = Qs + srow * 128 + sck;
#pragma unroll
    for (int i = 0; i < 8; i++)
      cp128_g2l(qsrc + (size_t)(16 * i) * HD, qdst + 16 * i * 128);
  }

  const unsigned short* kSrc = Kg + hb + (size_t)srow * HD + sck;
  unsigned short*       kDst = Ks + srow * 128 + sck;
  const unsigned short* vSrc = Vg + hb + (size_t)srow * HD + sck;

  float mrun[8], lrun[8];
  v8f o[8];
#pragma unroll
  for (int r = 0; r < 8; r++) { mrun[r] = -1e30f; lrun[r] = 0.0f; }
#pragma unroll
  for (int t = 0; t < 8; t++) o[t] = zero8();
  async_wait0();
  __syncthreads();

  for (int kb = 0; kb < L_ALL; kb += 32) {
    // K tile: 2 x b128 per thread, async
    cp128_g2l(kSrc + (size_t)kb * HD, kDst);
    cp128_g2l(kSrc + (size_t)(kb + 16) * HD, kDst + 16 * 128);
    // V tile transposed to [d][key] through regs
#pragma unroll
    for (int i = 0; i < 2; i++) {
      int key = srow + 16 * i;
      uint4 raw = *(const uint4*)(vSrc + (size_t)(kb + 16 * i) * HD);
      const unsigned short* e = (const unsigned short*)&raw;
#pragma unroll
      for (int jj = 0; jj < 8; jj++) Vs[(sck + jj) * 32 + key] = e[jj];
    }
    async_wait0();
    __syncthreads();

    // S = Q(16x128) * K^T(128x32) for this wave's rows
    v8f s0 = zero8(), s1 = zero8();
#pragma unroll
    for (int dk = 0; dk < 128; dk += 32) {
      v16bf aq = load_frag(Qs + (wave * 16) * 128 + dk, 128, lane);
      v16bf b0 = load_frag(Ks +              dk,        128, lane);   // keys 0..15
      v16bf b1 = load_frag(Ks + 16 * 128  +  dk,        128, lane);   // keys 16..31
      s0 = wmma_bf16(aq, b0, s0);
      s1 = wmma_bf16(aq, b1, s1);
    }

    // online softmax: row = r + 8*half, reductions across 16-lane halves
#pragma unroll
    for (int r = 0; r < 8; r++) {
      float mx = fmaxf(s0[r], s1[r]);
      for (int ofs = 1; ofs < 16; ofs <<= 1) mx = fmaxf(mx, __shfl_xor(mx, ofs, 32));
      float mnew = fmaxf(mrun[r], mx);
      float corr = __expf(mrun[r] - mnew);
      float p0 = __expf(s0[r] - mnew);
      float p1 = __expf(s1[r] - mnew);
      float rs = p0 + p1;
      for (int ofs = 1; ofs < 16; ofs <<= 1) rs += __shfl_xor(rs, ofs, 32);
      lrun[r] = lrun[r] * corr + rs;
      mrun[r] = mnew;
#pragma unroll
      for (int t = 0; t < 8; t++) o[t][r] = o[t][r] * corr;
      Ps[wave][(r + half * 8) * 32 + nl]      = f2bf(p0);
      Ps[wave][(r + half * 8) * 32 + 16 + nl] = f2bf(p1);
    }

    // O += P(16x32) * V(32x128)
    v16bf ap = load_frag(&Ps[wave][0], 32, lane);
#pragma unroll
    for (int t = 0; t < 8; t++) {
      v16bf bv = load_frag(Vs + (t * 16) * 32, 32, lane);
      o[t] = wmma_bf16(ap, bv, o[t]);
    }
    __syncthreads();   // before overwriting Ks/Vs
  }

  float inv[8];
#pragma unroll
  for (int r = 0; r < 8; r++) inv[r] = 1.0f / lrun[r];
#pragma unroll
  for (int t = 0; t < 8; t++) {
    int dcol = t * 16 + nl;
#pragma unroll
    for (int r = 0; r < 8; r++) {
      int row = q0 + wave * 16 + r + half * 8;
      Out[(size_t)row * HID + h * HD + dcol] = f2bf(o[t][r] * inv[r]);
    }
  }
}

// ---------------------------------------------------------------------------
// Host-side orchestration
// ---------------------------------------------------------------------------
extern "C" void kernel_launch(void* const* d_in, const int* in_sizes, int n_in,
                              void* d_out, int out_size, void* d_ws, size_t ws_size,
                              hipStream_t stream) {
  (void)in_sizes; (void)n_in; (void)out_size; (void)ws_size;
  const float* img        = (const float*)d_in[0];
  const float* txt        = (const float*)d_in[1];
  const float* vec        = (const float*)d_in[2];
  const float* pe         = (const float*)d_in[3];
  const float* img_mod_w  = (const float*)d_in[4];
  const float* img_mod_b  = (const float*)d_in[5];
  const float* img_qkv_w  = (const float*)d_in[6];
  const float* img_qkv_b  = (const float*)d_in[7];
  const float* img_q_scale= (const float*)d_in[8];
  const float* img_k_scale= (const float*)d_in[9];
  const float* img_proj_w = (const float*)d_in[10];
  const float* img_proj_b = (const float*)d_in[11];
  const float* img_mlp_w1 = (const float*)d_in[12];
  const float* img_mlp_b1 = (const float*)d_in[13];
  const float* img_mlp_w2 = (const float*)d_in[14];
  const float* img_mlp_b2 = (const float*)d_in[15];
  const float* txt_mod_w  = (const float*)d_in[16];
  const float* txt_mod_b  = (const float*)d_in[17];
  const float* txt_qkv_w  = (const float*)d_in[18];
  const float* txt_qkv_b  = (const float*)d_in[19];
  const float* txt_q_scale= (const float*)d_in[20];
  const float* txt_k_scale= (const float*)d_in[21];
  const float* txt_proj_w = (const float*)d_in[22];
  const float* txt_proj_b = (const float*)d_in[23];
  const float* txt_mlp_w1 = (const float*)d_in[24];
  const float* txt_mlp_b1 = (const float*)d_in[25];
  const float* txt_mlp_w2 = (const float*)d_in[26];
  const float* txt_mlp_b2 = (const float*)d_in[27];

  // workspace carving (deterministic, ~125 MB)
  char* wsp = (char*)d_ws;
  auto carve = [&](size_t bytes) -> void* {
    void* p = (void*)wsp;
    wsp += (bytes + 255) & ~(size_t)255;
    return p;
  };
  float* modImg = (float*)carve((size_t)NMOD * 4);
  float* modTxt = (float*)carve((size_t)NMOD * 4);
  unsigned short* lnImg = (unsigned short*)carve((size_t)L_IMG * HID * 2);
  unsigned short* lnTxt = (unsigned short*)carve((size_t)L_TXT * HID * 2);
  char* bigImg = (char*)carve((size_t)L_IMG * QKV_N * 4);  // qkv f32, later MLP-hid bf16
  char* bigTxt = (char*)carve((size_t)L_TXT * QKV_N * 4);
  unsigned short* Qb = (unsigned short*)carve((size_t)HEADS * L_ALL * HD * 2);
  unsigned short* Kb = (unsigned short*)carve((size_t)HEADS * L_ALL * HD * 2);
  unsigned short* Vb = (unsigned short*)carve((size_t)HEADS * L_ALL * HD * 2);
  unsigned short* attnB = (unsigned short*)carve((size_t)L_ALL * HID * 2);
  float* stateImg = (float*)carve((size_t)L_IMG * HID * 4);
  float* stateTxt = (float*)carve((size_t)L_TXT * HID * 4);

  // 1) modulation vectors (sh1,sc1,g1,sh2,sc2,g2) x 2048 each
  modulation_kernel<<<NMOD / 256, 256, 0, stream>>>(vec, img_mod_w, img_mod_b, modImg);
  modulation_kernel<<<NMOD / 256, 256, 0, stream>>>(vec, txt_mod_w, txt_mod_b, modTxt);

  // 2) LN1 + modulate -> bf16
  ln_mod_kernel<<<L_IMG, 256, 0, stream>>>(img, modImg + 0 * HID, modImg + 1 * HID, lnImg);
  ln_mod_kernel<<<L_TXT, 256, 0, stream>>>(txt, modTxt + 0 * HID, modTxt + 1 * HID, lnTxt);

  // 3) QKV GEMMs -> f32
  gemm_bf16w<<<dim3(QKV_N / 128, L_IMG / 128), 256, 0, stream>>>(
      lnImg, img_qkv_w, img_qkv_b, L_IMG, QKV_N, HID, EPI_F32,
      nullptr, nullptr, (float*)bigImg, nullptr);
  gemm_bf16w<<<dim3(QKV_N / 128, L_TXT / 128), 256, 0, stream>>>(
      lnTxt, txt_qkv_w, txt_qkv_b, L_TXT, QKV_N, HID, EPI_F32,
      nullptr, nullptr, (float*)bigTxt, nullptr);

  // 4) RMSNorm + RoPE + scale -> bf16 Q/K/V [H][L][D], txt tokens first
  qkv_prep_kernel<<<dim3(L_ALL, HEADS), 128, 0, stream>>>(
      (const float*)bigTxt, (const float*)bigImg,
      txt_q_scale, txt_k_scale, img_q_scale, img_k_scale, pe, Qb, Kb, Vb);

  // 5) flash attention -> bf16 [L][H*D]
  attn_kernel<<<dim3(L_ALL / 128, HEADS), 256, 0, stream>>>(Qb, Kb, Vb, attnB);

  // 6) output projection + gated residual -> f32 state
  gemm_bf16w<<<dim3(HID / 128, L_IMG / 128), 256, 0, stream>>>(
      attnB + (size_t)L_TXT * HID, img_proj_w, img_proj_b,
      L_IMG, HID, HID, EPI_RESGATE_F32, img, modImg + 2 * HID, stateImg, nullptr);
  gemm_bf16w<<<dim3(HID / 128, L_TXT / 128), 256, 0, stream>>>(
      attnB, txt_proj_w, txt_proj_b,
      L_TXT, HID, HID, EPI_RESGATE_F32, txt, modTxt + 2 * HID, stateTxt, nullptr);

  // 7) LN2 + modulate -> bf16 (reuse ln buffers)
  ln_mod_kernel<<<L_IMG, 256, 0, stream>>>(stateImg, modImg + 3 * HID, modImg + 4 * HID, lnImg);
  ln_mod_kernel<<<L_TXT, 256, 0, stream>>>(stateTxt, modTxt + 3 * HID, modTxt + 4 * HID, lnTxt);

  // 8) MLP up-proj + GELU -> bf16 hidden (reuse qkv buffers)
  unsigned short* hidImg = (unsigned short*)bigImg;
  unsigned short* hidTxt = (unsigned short*)bigTxt;
  gemm_bf16w<<<dim3(MLP_HID / 128, L_IMG / 128), 256, 0, stream>>>(
      lnImg, img_mlp_w1, img_mlp_b1, L_IMG, MLP_HID, HID, EPI_GELU_BF16,
      nullptr, nullptr, nullptr, hidImg);
  gemm_bf16w<<<dim3(MLP_HID / 128, L_TXT / 128), 256, 0, stream>>>(
      lnTxt, txt_mlp_w1, txt_mlp_b1, L_TXT, MLP_HID, HID, EPI_GELU_BF16,
      nullptr, nullptr, nullptr, hidTxt);

  // 9) MLP down-proj + gated residual -> final outputs (img then txt)
  float* outImg = (float*)d_out;
  float* outTxt = outImg + (size_t)L_IMG * HID;
  gemm_bf16w<<<dim3(HID / 128, L_IMG / 128), 256, 0, stream>>>(
      hidImg, img_mlp_w2, img_mlp_b2, L_IMG, HID, MLP_HID, EPI_RESGATE_F32,
      stateImg, modImg + 5 * HID, outImg, nullptr);
  gemm_bf16w<<<dim3(HID / 128, L_TXT / 128), 256, 0, stream>>>(
      hidTxt, txt_mlp_w2, txt_mlp_b2, L_TXT, HID, MLP_HID, EPI_RESGATE_F32,
      stateTxt, modTxt + 5 * HID, outTxt, nullptr);
}
